// TimeDecayConditionLayer_57389353009775
// MI455X (gfx1250) — matrix operand order repaired and checked by reading, not compiled, not run on hardware
//
#include <hip/hip_runtime.h>
#include <hip/hip_bf16.h>
#include <math.h>

// ---------------------------------------------------------------------------
// Sizes (fixed by the reference)
// ---------------------------------------------------------------------------
#define B_  4
#define T_  4096
#define E_  2048
#define H_  256
#define M_  (B_ * T_)          // 16384 tokens
#define N1_ (3 * H_)           // 768 fused projection columns (sc | xf | in_f)
#define NC_ 64                 // scan chunks per channel
#define CL_ (T_ / NC_)         // 64 steps per chunk

typedef __attribute__((ext_vector_type(16))) __bf16          v16bf;
typedef __attribute__((ext_vector_type(8)))  float           v8f;
typedef __attribute__((ext_vector_type(8)))  unsigned short  u16x8;

union BfVec { v16bf v; u16x8 h[2]; };

// ---------------------------------------------------------------------------
// Helpers
// ---------------------------------------------------------------------------
__device__ __forceinline__ float softplusf(float x) {
    return x > 20.0f ? x : log1pf(expf(x));
}

// Reduce two values across a 256-thread block (8 wave32 waves).
__device__ __forceinline__ void blk_reduce_2(float& s0, float& s1, float* sh) {
    const int lane = threadIdx.x & 31;
    const int w    = threadIdx.x >> 5;
#pragma unroll
    for (int o = 16; o > 0; o >>= 1) {
        s0 += __shfl_down(s0, o);
        s1 += __shfl_down(s1, o);
    }
    if (lane == 0) { sh[w] = s0; sh[8 + w] = s1; }
    __syncthreads();
    if (threadIdx.x == 0) {
        float a = 0.f, b = 0.f;
#pragma unroll
        for (int i = 0; i < 8; ++i) { a += sh[i]; b += sh[8 + i]; }
        sh[16] = a; sh[17] = b;
    }
    __syncthreads();
    s0 = sh[16]; s1 = sh[17];
    __syncthreads();
}

// Issue one async global->LDS 16B copy (CDNA5 GLOBAL_LOAD_ASYNC_TO_LDS_B128).
// lds_addr: 32-bit LDS byte address; byte_off: 32-bit byte offset from base.
__device__ __forceinline__ void async_b128(unsigned lds_addr, unsigned byte_off,
                                           const void* base) {
    asm volatile("global_load_async_to_lds_b128 %0, %1, %2"
                 :: "v"(lds_addr), "v"(byte_off), "s"(base) : "memory");
}

// ---------------------------------------------------------------------------
// K1: LayerNorm over E per token, emit bf16 activations
// ---------------------------------------------------------------------------
__global__ void k_ln_cast(const float* __restrict__ x,
                          const float* __restrict__ w,
                          const float* __restrict__ b,
                          __bf16* __restrict__ xln) {
    __shared__ float sh[18];
    const int m = blockIdx.x;
    const float* row = x + (size_t)m * E_;
    float4 v0 = ((const float4*)row)[threadIdx.x * 2 + 0];
    float4 v1 = ((const float4*)row)[threadIdx.x * 2 + 1];
    float vals[8] = {v0.x, v0.y, v0.z, v0.w, v1.x, v1.y, v1.z, v1.w};
    float s = 0.f, ss = 0.f;
#pragma unroll
    for (int i = 0; i < 8; ++i) { s += vals[i]; ss += vals[i] * vals[i]; }
    blk_reduce_2(s, ss, sh);
    const float mu   = s * (1.0f / E_);
    const float var  = ss * (1.0f / E_) - mu * mu;
    const float rstd = rsqrtf(var + 1e-5f);
    const int e0 = threadIdx.x * 8;
#pragma unroll
    for (int i = 0; i < 8; ++i) {
        float t = (vals[i] - mu) * rstd * w[e0 + i] + b[e0 + i];
        xln[(size_t)m * E_ + e0 + i] = (__bf16)t;
    }
}

// ---------------------------------------------------------------------------
// K2: fuse + cast the three projection matrices to bf16  WCAT[768][2048]
// ---------------------------------------------------------------------------
__global__ void k_cast_w(const float* __restrict__ sc_w,
                         const float* __restrict__ x_f_w,
                         const float* __restrict__ in_f_w,
                         __bf16* __restrict__ wcat) {
    const int n = blockIdx.x;
    const float* src;
    if (n < H_)          src = sc_w   + (size_t)n * E_;
    else if (n < 2 * H_) src = x_f_w  + (size_t)(n - H_) * E_;
    else                 src = in_f_w + (size_t)(n - 2 * H_) * E_;
    for (int k = threadIdx.x; k < E_; k += 256)
        wcat[(size_t)n * E_ + k] = (__bf16)src[k];
}

// K3: transpose+cast out_proj [H,E] -> OPT [E,H] bf16 (contiguous along K=h)
__global__ void k_cast_op(const float* __restrict__ op, __bf16* __restrict__ opt) {
    const int e = blockIdx.x;
    const int h = threadIdx.x;
    opt[(size_t)e * H_ + h] = (__bf16)op[(size_t)h * E_ + e];
}

// ---------------------------------------------------------------------------
// Tiled WMMA GEMM: block = 256 threads (8 waves) computes a 128x128 macro
// tile.  B panel (128 x 32) double-buffered in LDS via async global->LDS
// loads; each wave keeps 8 f32 accumulators (full 128-wide N coverage) and
// reuses its A fragment 8x from registers.
//   A: [M x K] bf16 row-major, B: [N x K] bf16 row-major (so B[k][n]=Bm[n][k])
// ---------------------------------------------------------------------------
template <int KTOT>
__device__ __forceinline__ void gemm_tile_mainloop(
        const __bf16* __restrict__ A, const __bf16* __restrict__ Bm,
        int m0, int n0, __bf16* Bs /* [2][128*32] LDS */, v8f acc[8]) {
    const int t    = threadIdx.x;
    const int lane = t & 31;
    const int w    = t >> 5;

    // staging geometry: 128 rows x 64B; thread t does rows t>>2 and (t>>2)+64
    const int srow = t >> 2;
    const int sseg = t & 3;
    const unsigned lds_base = (unsigned)(size_t)(void*)Bs;
    const unsigned lds_s0   = lds_base + (unsigned)(srow * 64 + sseg * 16);
    const unsigned goff0    = (unsigned)(((n0 + srow) * KTOT + sseg * 8) * 2);

    const __bf16* ap = A + (size_t)(m0 + w * 16 + (lane & 15)) * KTOT
                         + ((lane & 16) ? 8 : 0);
    const int koffB = (lane & 16) ? 8 : 0;
    const int nlow  = lane & 15;

    // prologue: stage first panel into buffer 0
    async_b128(lds_s0,                 goff0,                     Bm);
    async_b128(lds_s0 + 64u * 64u,     goff0 + 64u * KTOT * 2u,   Bm);

    for (int kk = 0; kk < KTOT; kk += 32) {
        const int  buf      = (kk >> 5) & 1;
        const bool havenext = (kk + 32 < KTOT);
        if (havenext) {
            const unsigned l = lds_base + (unsigned)((buf ^ 1) * 8192)
                             + (unsigned)(srow * 64 + sseg * 16);
            const unsigned g = goff0 + (unsigned)((kk + 32) * 2);
            async_b128(l,             g,                       Bm);
            async_b128(l + 64u * 64u, g + 64u * KTOT * 2u,     Bm);
            asm volatile("s_wait_asynccnt 2" ::: "memory");
        } else {
            asm volatile("s_wait_asynccnt 0" ::: "memory");
        }
        __syncthreads();   // current panel visible to all waves

        BfVec a;
        a.h[0] = *(const u16x8*)(ap + kk);
        a.h[1] = *(const u16x8*)(ap + kk + 16);

        const __bf16* bsrc = Bs + buf * (128 * 32);
#pragma unroll
        for (int j = 0; j < 8; ++j) {
            BfVec b;
            const __bf16* bp = bsrc + (j * 16 + nlow) * 32 + koffB;
            b.h[0] = *(const u16x8*)(bp);
            b.h[1] = *(const u16x8*)(bp + 16);
            acc[j] = __builtin_amdgcn_wmma_f32_16x16x32_bf16(
                false, a.v, false, b.v, (short)0, acc[j], false, false);
        }
        __syncthreads();   // all waves done reading before panel is overwritten
    }
}

// K4: GEMM1  P[M,768] = XLN[M,2048] x WCAT^T
__global__ void k_gemm1(const __bf16* __restrict__ A,
                        const __bf16* __restrict__ W,
                        float* __restrict__ C) {
    __shared__ __bf16 Bs[2][128 * 32];          // 16 KB double-buffered B panel
    const int NB = N1_ / 128;                   // 6
    const int m0 = (blockIdx.x / NB) * 128;
    const int n0 = (blockIdx.x % NB) * 128;
    v8f acc[8] = {};
    gemm_tile_mainloop<E_>(A, W, m0, n0, &Bs[0][0], acc);

    const int lane = threadIdx.x & 31;
    const int w    = threadIdx.x >> 5;
    const int mb   = m0 + w * 16 + ((lane & 16) ? 8 : 0);
#pragma unroll
    for (int j = 0; j < 8; ++j) {
        const int n = n0 + j * 16 + (lane & 15);
#pragma unroll
        for (int r = 0; r < 8; ++r)
            C[(size_t)(mb + r) * N1_ + n] = acc[j][r];
    }
}

// K10: GEMM2  out[M,2048] = x .* (YBF[M,256] x OPT^T)
__global__ void k_gemm2(const __bf16* __restrict__ A,
                        const __bf16* __restrict__ Bm,
                        const float* __restrict__ x,
                        float* __restrict__ out) {
    __shared__ __bf16 Bs[2][128 * 32];
    const int NB = E_ / 128;                    // 16
    const int m0 = (blockIdx.x / NB) * 128;
    const int n0 = (blockIdx.x % NB) * 128;
    v8f acc[8] = {};
    gemm_tile_mainloop<H_>(A, Bm, m0, n0, &Bs[0][0], acc);

    const int lane = threadIdx.x & 31;
    const int w    = threadIdx.x >> 5;
    const int mb   = m0 + w * 16 + ((lane & 16) ? 8 : 0);
#pragma unroll
    for (int j = 0; j < 8; ++j) {
        const int n = n0 + j * 16 + (lane & 15);
#pragma unroll
        for (int r = 0; r < 8; ++r) {
            const size_t idx = (size_t)(mb + r) * E_ + n;
            out[idx] = x[idx] * acc[j][r];
        }
    }
}

// ---------------------------------------------------------------------------
// K5: per-token epilogue: sc gate (softplus + L2-norm * sqrt(H)), xf bias,
//     a = -softplus(.)   (all in place in P)
// ---------------------------------------------------------------------------
__global__ void k_act(float* __restrict__ P,
                      const float* __restrict__ sc_b,
                      const float* __restrict__ xf_b,
                      const float* __restrict__ inf_b) {
    __shared__ float sh[18];
    const int m = blockIdx.x, h = threadIdx.x;
    float* row = P + (size_t)m * N1_;
    float s = softplusf(row[h] + sc_b[h]) + 1e-8f;
    float ss = s * s, dummy = 0.f;
    blk_reduce_2(ss, dummy, sh);
    const float nrm = sqrtf(ss);
    const float scv = s / fmaxf(nrm, 1e-8f) * 16.0f;      // sqrt(H)=16
    const float xfv = row[H_ + h] + xf_b[h];
    const float av  = -softplusf(row[2 * H_ + h] + inf_b[h]);
    row[h] = scv; row[H_ + h] = xfv; row[2 * H_ + h] = av;
}

// ---------------------------------------------------------------------------
// K6-K8: chunked linear recurrence  y_t = exp(a_t)*y_{t-1} + xf_t
// ---------------------------------------------------------------------------
__global__ void k_scan1(const float* __restrict__ P, float* __restrict__ Y,
                        float* __restrict__ CH) {
    const int tid = blockIdx.x * 256 + threadIdx.x;     // 65536 threads
    const int h = tid & 255;
    const int c = (tid >> 8) & (NC_ - 1);
    const int b = tid >> 14;
    float y = 0.f, asum = 0.f;
    const int t0 = c * CL_;
#pragma unroll 4
    for (int t = t0; t < t0 + CL_; ++t) {
        const size_t ridx = ((size_t)b * T_ + t) * N1_;
        const float a  = P[ridx + 2 * H_ + h];
        const float xf = P[ridx + H_ + h];
        asum += a;
        y = expf(a) * y + xf;
        Y[((size_t)b * T_ + t) * H_ + h] = y;
    }
    const size_t ch = (size_t)(b * H_ + h) * NC_ + c;
    CH[ch * 2 + 0] = expf(asum);
    CH[ch * 2 + 1] = y;
}

__global__ void k_scan2(const float* __restrict__ CH, float* __restrict__ CARRY) {
    const int ch = blockIdx.x * 256 + threadIdx.x;      // 1024 channels
    float carry = 0.f;
#pragma unroll 4
    for (int c = 0; c < NC_; ++c) {
        const size_t i = (size_t)ch * NC_ + c;
        CARRY[i] = carry;
        carry = CH[i * 2 + 0] * carry + CH[i * 2 + 1];
    }
}

__global__ void k_scan3(const float* __restrict__ P, float* __restrict__ Y,
                        const float* __restrict__ CARRY) {
    const int tid = blockIdx.x * 256 + threadIdx.x;
    const int h = tid & 255;
    const int c = (tid >> 8) & (NC_ - 1);
    const int b = tid >> 14;
    const float carry = CARRY[(size_t)(b * H_ + h) * NC_ + c];
    float asum = 0.f;
    const int t0 = c * CL_;
#pragma unroll 4
    for (int t = t0; t < t0 + CL_; ++t) {
        const float a = P[((size_t)b * T_ + t) * N1_ + 2 * H_ + h];
        asum += a;
        Y[((size_t)b * T_ + t) * H_ + h] += expf(asum) * carry;
    }
}

// ---------------------------------------------------------------------------
// K9: second LayerNorm over H, gated by sc, cast to bf16
// ---------------------------------------------------------------------------
__global__ void k_ln2(const float* __restrict__ Y, const float* __restrict__ P,
                      const float* __restrict__ ln_w, const float* __restrict__ ln_b,
                      __bf16* __restrict__ ybf) {
    __shared__ float sh[18];
    const int m = blockIdx.x, h = threadIdx.x;
    const float v = Y[(size_t)m * H_ + h];
    float s = v, ss = v * v;
    blk_reduce_2(s, ss, sh);
    const float mu   = s * (1.0f / H_);
    const float var  = ss * (1.0f / H_) - mu * mu;
    const float rstd = rsqrtf(var + 1e-5f);
    const float val  = (v - mu) * rstd * ln_w[h] + ln_b[h];
    const float scv  = P[(size_t)m * N1_ + h];
    ybf[(size_t)m * H_ + h] = (__bf16)(scv * val);
}

// ---------------------------------------------------------------------------
// Host launcher
// ---------------------------------------------------------------------------
extern "C" void kernel_launch(void* const* d_in, const int* in_sizes, int n_in,
                              void* d_out, int out_size, void* d_ws, size_t ws_size,
                              hipStream_t stream) {
    const float* x      = (const float*)d_in[0];
    const float* norm_w = (const float*)d_in[1];
    const float* norm_b = (const float*)d_in[2];
    const float* in_f_w = (const float*)d_in[3];
    const float* in_f_b = (const float*)d_in[4];
    const float* x_f_w  = (const float*)d_in[5];
    const float* x_f_b  = (const float*)d_in[6];
    const float* sc_w   = (const float*)d_in[7];
    const float* sc_b   = (const float*)d_in[8];
    const float* ln_w   = (const float*)d_in[9];
    const float* ln_b   = (const float*)d_in[10];
    const float* o_proj = (const float*)d_in[11];
    float* out = (float*)d_out;

    // Workspace carve-up (all offsets 256B-aligned)
    char* w = (char*)d_ws;
    size_t off = 0;
    __bf16* XLN  = (__bf16*)(w + off); off += (size_t)M_ * E_ * 2;        // 64 MB
    __bf16* WCAT = (__bf16*)(w + off); off += (size_t)N1_ * E_ * 2;       // 3 MB
    __bf16* OPT  = (__bf16*)(w + off); off += (size_t)E_ * H_ * 2;        // 1 MB
    float*  P    = (float*) (w + off); off += (size_t)M_ * N1_ * 4;       // 48 MB
    float*  Y    = (float*) (w + off); off += (size_t)M_ * H_ * 4;        // 16 MB
    __bf16* YBF  = (__bf16*)(w + off); off += (size_t)M_ * H_ * 2;        // 8 MB
    float*  CH   = (float*) (w + off); off += (size_t)B_ * H_ * NC_ * 2 * 4;
    float*  CAR  = (float*) (w + off); off += (size_t)B_ * H_ * NC_ * 4;

    // 1. LayerNorm(E) -> bf16 activations
    k_ln_cast<<<M_, 256, 0, stream>>>(x, norm_w, norm_b, XLN);
    // 2-3. weight casts
    k_cast_w <<<N1_, 256, 0, stream>>>(sc_w, x_f_w, in_f_w, WCAT);
    k_cast_op<<<E_,  256, 0, stream>>>(o_proj, OPT);
    // 4. fused triple projection GEMM: 128x128 macro tiles, async-LDS B panels
    k_gemm1<<<(M_ / 128) * (N1_ / 128), 256, 0, stream>>>(XLN, WCAT, P);
    // 5. epilogues (softplus gate + L2 norm, bias, -softplus)
    k_act<<<M_, 256, 0, stream>>>(P, sc_b, x_f_b, in_f_b);
    // 6-8. chunked linear-recurrence scan over T
    k_scan1<<<(B_ * H_ * NC_) / 256, 256, 0, stream>>>(P, Y, CH);
    k_scan2<<<(B_ * H_) / 256,       256, 0, stream>>>(CH, CAR);
    k_scan3<<<(B_ * H_ * NC_) / 256, 256, 0, stream>>>(P, Y, CAR);
    // 9. LayerNorm(H) * gate -> bf16
    k_ln2<<<M_, 256, 0, stream>>>(Y, P, ln_w, ln_b, YBF);
    // 10. output GEMM fused with x * (.)
    k_gemm2<<<(M_ / 128) * (E_ / 128), 256, 0, stream>>>(YBF, OPT, x, out);
    (void)in_sizes; (void)n_in; (void)out_size; (void)ws_size; (void)off;
}
